// DepthawareConv_40905268527560
// MI455X (gfx1250) — matrix operand order repaired and checked by reading, not compiled, not run on hardware
//
#include <hip/hip_runtime.h>
#include <hip/hip_bf16.h>

typedef __attribute__((ext_vector_type(16))) _Float16     v16h;
typedef __attribute__((ext_vector_type(8)))  float        v8f;
typedef __attribute__((ext_vector_type(4)))  unsigned int u32x4;
typedef __attribute__((ext_vector_type(8)))  int          i32x8;
typedef __attribute__((ext_vector_type(4)))  int          i32x4;

#define ALPHA 8.3f
#define CIN   32
#define COUT  32
#define IMH   256
#define IMW   256
#define NB    4
#define TILE_H 8
#define TILE_W 16
#define XR (TILE_H + 2)   // 10 rows incl. halo
#define XC (TILE_W + 2)   // 18 cols incl. halo
#define CP 36             // padded channel stride (keeps 16B-aligned c0=0/16 bases)
#define NWFRAG (9 * 2 * 32 * 16)   // 9216 halfs = 18 KB of swizzled A-fragments

// ---------------------------------------------------------------------------
// Prep kernel: swizzle weights into this kernel's contiguous-k A-fragment lane order, f16,
// into d_ws. Runs once per launch (1 WG); main kernel TDM-loads this block.
//   lane L<16 : slots 0..15 -> K=0..15
//   lane L>=16: slots 0..15 -> K=16..31
// ---------------------------------------------------------------------------
__global__ __launch_bounds__(256) void prep_weights_kernel(
    const float* __restrict__ weight, _Float16* __restrict__ wfrag)
{
    for (int idx = threadIdx.x; idx < NWFRAG; idx += 256) {
        int s = idx & 15;
        int L = (idx >> 4) & 31;
        int m = (idx >> 9) & 1;
        int t = idx >> 10;                       // tap 0..8
        int o = m * 16 + (L & 15);
        int c = (L & 16) + s;
        wfrag[idx] = (_Float16)weight[(o * CIN + c) * 9 + t];
    }
}

__global__ __launch_bounds__(256) void dconv_wmma_kernel(
    const float* __restrict__ x, const float* __restrict__ depth,
    const _Float16* __restrict__ wfrag, float* __restrict__ out)
{
    __shared__ __align__(16) float    ldsX[XR * XC * CP];    // x tile, [row][col][c]
    __shared__            float       ldsD[XR * XC];         // depth tile
    __shared__ __align__(32) _Float16 ldsW[NWFRAG];          // A-frags via TDM

    const int tid = threadIdx.x;
    const int b   = blockIdx.z;
    const int h0  = blockIdx.y * TILE_H;
    const int w0  = blockIdx.x * TILE_W;

    // ---- TDM: DMA the 18KB pre-swizzled weight block into LDS (wave 0) ----
    if (tid < 32) {
        unsigned ldsOff = (unsigned)(uintptr_t)ldsW;                 // LDS byte offset
        unsigned long long ga = (unsigned long long)(uintptr_t)wfrag;
        u32x4 g0;
        g0[0] = 1u;                                // count=1, user descriptor
        g0[1] = ldsOff;                            // lds_addr
        g0[2] = (unsigned)ga;                      // global_addr[31:0]
        g0[3] = (unsigned)(ga >> 32) | (2u << 30); // global_addr[56:32] | type=2
        i32x8 g1;
        g1[0] = 0x20000;       // workgroup_mask=0, data_size=2 (4B)
        g1[1] = 0x12000000;    // tensor_dim0[15:0]=4608 in bits[31:16]
        g1[2] = 0x00010000;    // tensor_dim0[31:16]=0, tensor_dim1[15:0]=1
        g1[3] = 0x12000000;    // tensor_dim1 hi=0, tile_dim0=4608
        g1[4] = 1;             // tile_dim1=1, tile_dim2=0
        g1[5] = 4608;          // tensor_dim0_stride[31:0]
        g1[6] = 0x12000000;    // stride0 hi=0, tensor_dim1_stride[15:0]=4608
        g1[7] = 0;
        i32x4 g2 = {0, 0, 0, 0};
        i32x4 g3 = {0, 0, 0, 0};
#if __clang_major__ >= 23
        i32x8 g4 = {0, 0, 0, 0, 0, 0, 0, 0};
        __builtin_amdgcn_tensor_load_to_lds(g0, g1, g2, g3, g4, 0);
#else
        __builtin_amdgcn_tensor_load_to_lds(g0, g1, g2, g3, 0);
#endif
    }

    // ---- stage depth halo tile (OOB -> 0; x is 0 there so sim irrelevant) ----
    for (int idx = tid; idx < XR * XC; idx += 256) {
        int col = idx % XC, r = idx / XC;
        int h = h0 - 1 + r, w = w0 - 1 + col;
        float v = 0.f;
        if (h >= 0 && h < IMH && w >= 0 && w < IMW)
            v = depth[(b * IMH + h) * IMW + w];
        ldsD[idx] = v;
    }

    // ---- stage x halo tile transposed to [row][col][c] ----
    const bool interior = (h0 >= 1) && (h0 + TILE_H < IMH) &&
                          (w0 >= 1) && (w0 + TILE_W < IMW);
    if (interior) {
        // Fast path: async global->LDS copies; per-lane global addr + per-lane
        // LDS addr performs the [c][h][w] -> [row][col][c] transpose in flight.
        const unsigned ldsXOff = (unsigned)(uintptr_t)ldsX;
        for (int idx = tid; idx < CIN * XR * XC; idx += 256) {
            int col = idx % XC;
            int r   = (idx / XC) % XR;
            int c   = idx / (XR * XC);
            unsigned gOff = (unsigned)(((((b * CIN + c) * IMH + (h0 - 1 + r)) * IMW)
                                        + (w0 - 1 + col)) * 4);
            unsigned lOff = ldsXOff + (unsigned)(((r * XC + col) * CP + c) * 4);
            asm volatile("global_load_async_to_lds_b32 %0, %1, %2"
                         :: "v"(lOff), "v"(gOff), "s"(x) : "memory");
        }
        asm volatile("s_wait_asynccnt 0x0" ::: "memory");
    } else {
        for (int idx = tid; idx < CIN * XR * XC; idx += 256) {
            int col = idx % XC;
            int r   = (idx / XC) % XR;
            int c   = idx / (XR * XC);
            int h = h0 - 1 + r, w = w0 - 1 + col;
            float v = 0.f;
            if (h >= 0 && h < IMH && w >= 0 && w < IMW)
                v = x[((b * CIN + c) * IMH + h) * IMW + w];
            ldsX[(r * XC + col) * CP + c] = v;
        }
    }

    if (tid < 32) __builtin_amdgcn_s_wait_tensorcnt(0);
    __syncthreads();

    const int lane = tid & 31;
    const int wave = tid >> 5;          // tile row this wave computes
    const int p    = lane & 15;         // pixel column within tile (N index)
    const int c0   = lane & 16;         // B-frag channel base: 0 or 16

    const float dc = ldsD[(wave + 1) * XC + (p + 1)];

    v8f acc0 = {}, acc1 = {};

    #pragma unroll
    for (int t = 0; t < 9; ++t) {
        const int dh = t / 3 - 1, dw = t % 3 - 1;
        const float dt  = ldsD[(wave + 1 + dh) * XC + (p + 1 + dw)];
        const float sim = __expf(-ALPHA * fabsf(dc - dt));

        // B fragment: 32x16 f16, lane holds 16 consecutive channels of one pixel
        const float* xp = &ldsX[((wave + 1 + dh) * XC + (p + 1 + dw)) * CP + c0];
        v16h bfrag;
        #pragma unroll
        for (int s = 0; s < 16; ++s)
            bfrag[s] = (_Float16)(xp[s] * sim);

        // A fragments for both 16-row output-channel tiles (TDM-loaded)
        v16h a0 = *(const v16h*)(ldsW + ((t * 2 + 0) * 32 + lane) * 16);
        v16h a1 = *(const v16h*)(ldsW + ((t * 2 + 1) * 32 + lane) * 16);

        acc0 = __builtin_amdgcn_wmma_f32_16x16x32_f16(
            false, a0, false, bfrag, (short)0, acc0, false, false);
        acc1 = __builtin_amdgcn_wmma_f32_16x16x32_f16(
            false, a1, false, bfrag, (short)0, acc1, false, false);
    }

    // ---- store: D layout: VGPR r -> M=r (lanes 0-15) / M=8+r (lanes 16-31), N=p
    const int h = h0 + wave, w = w0 + p;
    const int oBase = (lane < 16) ? 0 : 8;
    #pragma unroll
    for (int r = 0; r < 8; ++r) {
        const int o = oBase + r;
        out[((b * COUT + o)        * IMH + h) * IMW + w] = acc0[r];
        out[((b * COUT + (16 + o)) * IMH + h) * IMW + w] = acc1[r];
    }
}

extern "C" void kernel_launch(void* const* d_in, const int* in_sizes, int n_in,
                              void* d_out, int out_size, void* d_ws, size_t ws_size,
                              hipStream_t stream) {
    (void)in_sizes; (void)n_in; (void)out_size; (void)ws_size;
    const float* x      = (const float*)d_in[0];
    const float* depth  = (const float*)d_in[1];
    const float* weight = (const float*)d_in[2];
    float* out          = (float*)d_out;
    _Float16* wfrag     = (_Float16*)d_ws;

    prep_weights_kernel<<<1, 256, 0, stream>>>(weight, wfrag);

    dim3 grid(IMW / TILE_W, IMH / TILE_H, NB);   // 16 x 32 x 4 = 2048 WGs
    dconv_wmma_kernel<<<grid, 256, 0, stream>>>(x, depth, wfrag, out);
}
